// MixtralSparseMoeBlock_21182778704249
// MI455X (gfx1250) — matrix operand order repaired and checked by reading, not compiled, not run on hardware
//
#include <hip/hip_runtime.h>
#include <hip/hip_bf16.h>

// Mixtral sparse MoE block for gfx1250 (MI455X), bf16 WMMA path.
// T=4096 tokens, H=1024 hidden, I=3584 intermediate, E=8 experts, top-2.
// Token-gathered grouped GEMMs (only selected experts computed),
// double-buffered LDS + register-prefetch software pipeline.

#define T_TOK 4096
#define HDIM  1024
#define IDIM  3584
#define NEXP  8

#define BM 128
#define BN 128
#define BK 32
#define LDA 48   // ushort row stride in LDS (96 bytes, 16B aligned)

typedef __attribute__((ext_vector_type(16))) __bf16 bf16x16;
typedef __attribute__((ext_vector_type(8)))  __bf16 bf16x8;
typedef __attribute__((ext_vector_type(2)))  __bf16 bf16x2;
typedef __attribute__((ext_vector_type(8)))  float  v8f;

// ---------- bf16 conversion helpers ----------

__device__ __forceinline__ unsigned short f2bf(float v) {
  unsigned u = __float_as_uint(v);
  return (unsigned short)((u + 0x7FFFu + ((u >> 16) & 1u)) >> 16);  // RNE
}

#if __has_builtin(__builtin_amdgcn_cvt_pk_bf16_f32)
__device__ __forceinline__ unsigned f2bf_pack(float a, float b) {
  bf16x2 v = __builtin_amdgcn_cvt_pk_bf16_f32(a, b);  // v_cvt_pk_bf16_f32
  union { bf16x2 v; unsigned u; } c;
  c.v = v;
  return c.u;
}
#else
__device__ __forceinline__ unsigned f2bf_pack(float a, float b) {
  return (unsigned)f2bf(a) | ((unsigned)f2bf(b) << 16);
}
#endif

// ---------- tile staging helpers ----------

// 16 consecutive fp32 from global -> 4 float4 regs
__device__ __forceinline__ void gload16(float4* r, const float* src) {
  r[0] = ((const float4*)src)[0];
  r[1] = ((const float4*)src)[1];
  r[2] = ((const float4*)src)[2];
  r[3] = ((const float4*)src)[3];
}

// 4 float4 regs -> 16 bf16 in LDS (two b128 stores)
__device__ __forceinline__ void sstore16(unsigned short* dst, const float4* r) {
  uint4 u0 = { f2bf_pack(r[0].x, r[0].y), f2bf_pack(r[0].z, r[0].w),
               f2bf_pack(r[1].x, r[1].y), f2bf_pack(r[1].z, r[1].w) };
  uint4 u1 = { f2bf_pack(r[2].x, r[2].y), f2bf_pack(r[2].z, r[2].w),
               f2bf_pack(r[3].x, r[3].y), f2bf_pack(r[3].z, r[3].w) };
  ((uint4*)dst)[0] = u0;
  ((uint4*)dst)[1] = u1;
}

// two 16B LDS reads -> one 16-element bf16 fragment
__device__ __forceinline__ bf16x16 ld_frag(const unsigned short* p0,
                                           const unsigned short* p1) {
  bf16x8 lo = *(const bf16x8*)p0;
  bf16x8 hi = *(const bf16x8*)p1;
  return __builtin_shufflevector(lo, hi, 0,1,2,3,4,5,6,7,8,9,10,11,12,13,14,15);
}

__device__ __forceinline__ v8f wmma_bf16(bf16x16 a, bf16x16 b, v8f c) {
  return __builtin_amdgcn_wmma_f32_16x16x32_bf16(
      /*neg_a=*/false, a, /*neg_b=*/false, b,
      /*c_mod=*/(short)0, c, /*reuse_a=*/false, /*reuse_b=*/false);
}

__device__ __forceinline__ v8f v8zero() {
  v8f z = {0.f, 0.f, 0.f, 0.f, 0.f, 0.f, 0.f, 0.f};
  return z;
}

// ---------- kernel 0: zero output + expert counters ----------

__global__ __launch_bounds__(256) void moe_zero_kernel(float* __restrict__ out,
                                                       int* __restrict__ cnt) {
  size_t idx = (size_t)blockIdx.x * 256 + threadIdx.x;
  float4 z = {0.f, 0.f, 0.f, 0.f};
  ((float4*)out)[idx] = z;
  if (blockIdx.x == 0 && threadIdx.x < NEXP) cnt[threadIdx.x] = 0;
}

// ---------- kernel 1: router (softmax + top-2 + gather lists) ----------

__global__ __launch_bounds__(256) void moe_router_kernel(
    const float* __restrict__ x, const float* __restrict__ gate_w,
    int* __restrict__ cnt, int* __restrict__ list, float* __restrict__ wgt) {
  __shared__ float gw[NEXP * HDIM];  // 32 KB: whole gate matrix
  const int tid = threadIdx.x;
  for (int i = tid; i < NEXP * HDIM; i += 256) gw[i] = gate_w[i];
  __syncthreads();

  const int wv = tid >> 5, lane = tid & 31;
  const int t = blockIdx.x * 8 + wv;  // one wave32 per token
  const float* xr = x + (size_t)t * HDIM;

  float a[NEXP];
#pragma unroll
  for (int e = 0; e < NEXP; ++e) a[e] = 0.f;
  for (int k = 0; k < HDIM; k += 32) {
    float xv = xr[k + lane];
#pragma unroll
    for (int e = 0; e < NEXP; ++e) a[e] += xv * gw[e * HDIM + k + lane];
  }
#pragma unroll
  for (int e = 0; e < NEXP; ++e) {
#pragma unroll
    for (int off = 16; off > 0; off >>= 1) a[e] += __shfl_xor(a[e], off, 32);
  }

  if (lane == 0) {
    float mx = a[0];
#pragma unroll
    for (int e = 1; e < NEXP; ++e) mx = fmaxf(mx, a[e]);
    float p[NEXP];
#pragma unroll
    for (int e = 0; e < NEXP; ++e) p[e] = __expf(a[e] - mx);
    int i1 = 0;
#pragma unroll
    for (int e = 1; e < NEXP; ++e) if (p[e] > p[i1]) i1 = e;
    int i2 = (i1 == 0) ? 1 : 0;
#pragma unroll
    for (int e = 0; e < NEXP; ++e) if (e != i1 && p[e] > p[i2]) i2 = e;
    float inv = 1.f / (p[i1] + p[i2]);
    int q1 = atomicAdd(&cnt[i1], 1);
    list[i1 * T_TOK + q1] = t;
    wgt[i1 * T_TOK + q1] = p[i1] * inv;
    int q2 = atomicAdd(&cnt[i2], 1);
    list[i2 * T_TOK + q2] = t;
    wgt[i2 * T_TOK + q2] = p[i2] * inv;
  }
}

// ---------- kernel 2: up projection (w1 & w3 fused) + SwiGLU ----------
// gathered tokens for expert e; act[pos, I] (bf16) = silu(x@w1^T) * (x@w3^T)

__global__ __launch_bounds__(256) void moe_up_kernel(
    const float* __restrict__ x, const float* __restrict__ w1,
    const float* __restrict__ w3, const int* __restrict__ cnt,
    const int* __restrict__ list, unsigned short* __restrict__ act, int e) {
  const int c = cnt[e];
  const int mbase = blockIdx.x * BM;
  if (mbase >= c) return;  // dynamic early-exit (worst-case grid)
  const int nbase = blockIdx.y * BN;

  __shared__ unsigned short As[2][BM * LDA];   // 24 KB
  __shared__ unsigned short B1s[2][BN * LDA];  // 24 KB
  __shared__ unsigned short B3s[2][BN * LDA];  // 24 KB

  const int tid = threadIdx.x;
  const int row = tid >> 1;
  const int seg = (tid & 1) * 16;
  const int pos = mbase + row;
  const int gp = pos < c ? pos : c - 1;
  const int tok = list[e * T_TOK + gp];
  const float* xrow  = x  + (size_t)tok * HDIM + seg;
  const float* w1row = w1 + ((size_t)e * IDIM + (nbase + row)) * HDIM + seg;
  const float* w3row = w3 + ((size_t)e * IDIM + (nbase + row)) * HDIM + seg;
  const int soff = row * LDA + seg;

  const int lane = tid & 31;
  const int wv = tid >> 5;
  const int waveM = wv & 3;   // 4 waves x 32 rows = 128
  const int waveN = wv >> 2;  // 2 waves x 64 cols = 128
  const int l16 = lane & 15;
  const int half = lane >> 4;

  v8f acc1[2][4], acc3[2][4];
#pragma unroll
  for (int mf = 0; mf < 2; ++mf)
#pragma unroll
    for (int nf = 0; nf < 4; ++nf) { acc1[mf][nf] = v8zero(); acc3[mf][nf] = v8zero(); }

  // prologue: stage tile 0 into buffer 0
  float4 ra[4], rb1[4], rb3[4];
  gload16(ra,  xrow);
  gload16(rb1, w1row);
  gload16(rb3, w3row);
  sstore16(&As[0][soff],  ra);
  sstore16(&B1s[0][soff], rb1);
  sstore16(&B3s[0][soff], rb3);
  __syncthreads();

  int pb = 0;
  for (int kk = 0; kk < HDIM; kk += BK) {
    const bool more = (kk + BK) < HDIM;
    if (more) {
      // issue next tile's global loads before the MMA chain (latency overlap)
      gload16(ra,  xrow  + kk + BK);
      gload16(rb1, w1row + kk + BK);
      gload16(rb3, w3row + kk + BK);
      __builtin_prefetch(w1row + kk + 2 * BK, 0, 0);
      __builtin_prefetch(w3row + kk + 2 * BK, 0, 0);
    }

    // compute on resident buffer pb
    bf16x16 af[2];
#pragma unroll
    for (int mf = 0; mf < 2; ++mf) {
      const unsigned short* ab = &As[pb][(waveM * 32 + mf * 16 + l16) * LDA];
      af[mf] = ld_frag(ab + 8 * half, ab + 8 * half + 16);
    }
#pragma unroll
    for (int nf = 0; nf < 4; ++nf) {
      const unsigned short* b1b = &B1s[pb][(waveN * 64 + nf * 16 + l16) * LDA + 8 * half];
      bf16x16 bf1 = ld_frag(b1b, b1b + 16);
      acc1[0][nf] = wmma_bf16(af[0], bf1, acc1[0][nf]);
      acc1[1][nf] = wmma_bf16(af[1], bf1, acc1[1][nf]);
      const unsigned short* b3b = &B3s[pb][(waveN * 64 + nf * 16 + l16) * LDA + 8 * half];
      bf16x16 bf3 = ld_frag(b3b, b3b + 16);
      acc3[0][nf] = wmma_bf16(af[0], bf3, acc3[0][nf]);
      acc3[1][nf] = wmma_bf16(af[1], bf3, acc3[1][nf]);
    }

    if (more) {
      // convert + park next tile in the alternate buffer
      sstore16(&As[pb ^ 1][soff],  ra);
      sstore16(&B1s[pb ^ 1][soff], rb1);
      sstore16(&B3s[pb ^ 1][soff], rb3);
    }
    __syncthreads();  // single barrier per K-step
    pb ^= 1;
  }

  // fused SwiGLU epilogue -> bf16 activations
#pragma unroll
  for (int mf = 0; mf < 2; ++mf) {
#pragma unroll
    for (int nf = 0; nf < 4; ++nf) {
      v8f g = acc1[mf][nf], u = acc3[mf][nf];
      const int coln = nbase + waveN * 64 + nf * 16 + l16;
#pragma unroll
      for (int r = 0; r < 8; ++r) {
        const int lm = waveM * 32 + mf * 16 + half * 8 + r;
        const int pm = mbase + lm;
        if (pm < c) {
          float gv = g[r];
          float sv = gv / (1.f + __expf(-gv)) * u[r];  // silu(g) * u
          act[(size_t)pm * IDIM + coln] = f2bf(sv);
        }
      }
    }
  }
}

// ---------- kernel 3: down projection + weighted scatter-add ----------

__global__ __launch_bounds__(256) void moe_down_kernel(
    const float* __restrict__ w2, const int* __restrict__ cnt,
    const int* __restrict__ list, const float* __restrict__ wgt,
    const unsigned short* __restrict__ act, float* __restrict__ out, int e) {
  const int c = cnt[e];
  const int mbase = blockIdx.x * BM;
  if (mbase >= c) return;
  const int hbase = blockIdx.y * BN;

  __shared__ unsigned short As[2][BM * LDA];  // 24 KB (bf16 activations)
  __shared__ unsigned short Bs[2][BN * LDA];  // 24 KB (converted w2)
  __shared__ int   toksh[BM];
  __shared__ float wtsh[BM];

  const int tid = threadIdx.x;
  if (tid < BM) {
    int p = mbase + tid;
    int gp = p < c ? p : c - 1;
    toksh[tid] = list[e * T_TOK + gp];
    wtsh[tid] = (p < c) ? wgt[e * T_TOK + gp] : 0.f;
  }

  const int row = tid >> 1;
  const int seg = (tid & 1) * 16;
  const int p0 = mbase + row;
  const int gp0 = p0 < c ? p0 : c - 1;
  const unsigned short* arow = act + (size_t)gp0 * IDIM + seg;
  const float* w2row = w2 + ((size_t)e * HDIM + (hbase + row)) * IDIM + seg;
  const int soff = row * LDA + seg;

  const int lane = tid & 31;
  const int wv = tid >> 5;
  const int waveM = wv & 3;
  const int waveN = wv >> 2;
  const int l16 = lane & 15;
  const int half = lane >> 4;

  v8f acc[2][4];
#pragma unroll
  for (int mf = 0; mf < 2; ++mf)
#pragma unroll
    for (int nf = 0; nf < 4; ++nf) acc[mf][nf] = v8zero();

  // prologue: tile 0 -> buffer 0
  uint4 rau[2];
  float4 rbw[4];
  rau[0] = ((const uint4*)arow)[0];
  rau[1] = ((const uint4*)arow)[1];
  gload16(rbw, w2row);
  ((uint4*)&As[0][soff])[0] = rau[0];
  ((uint4*)&As[0][soff])[1] = rau[1];
  sstore16(&Bs[0][soff], rbw);
  __syncthreads();

  int pb = 0;
  for (int kk = 0; kk < IDIM; kk += BK) {
    const bool more = (kk + BK) < IDIM;
    if (more) {
      const uint4* s = (const uint4*)(arow + kk + BK);
      rau[0] = s[0];
      rau[1] = s[1];
      gload16(rbw, w2row + kk + BK);
      __builtin_prefetch(w2row + kk + 2 * BK, 0, 0);
    }

    bf16x16 af[2];
#pragma unroll
    for (int mf = 0; mf < 2; ++mf) {
      const unsigned short* ab = &As[pb][(waveM * 32 + mf * 16 + l16) * LDA];
      af[mf] = ld_frag(ab + 8 * half, ab + 8 * half + 16);
    }
#pragma unroll
    for (int nf = 0; nf < 4; ++nf) {
      const unsigned short* bb = &Bs[pb][(waveN * 64 + nf * 16 + l16) * LDA + 8 * half];
      bf16x16 bfm = ld_frag(bb, bb + 16);
      acc[0][nf] = wmma_bf16(af[0], bfm, acc[0][nf]);
      acc[1][nf] = wmma_bf16(af[1], bfm, acc[1][nf]);
    }

    if (more) {
      ((uint4*)&As[pb ^ 1][soff])[0] = rau[0];
      ((uint4*)&As[pb ^ 1][soff])[1] = rau[1];
      sstore16(&Bs[pb ^ 1][soff], rbw);
    }
    __syncthreads();
    pb ^= 1;
  }

  // weighted combine: out[tok, h] += w * acc  (each token hit by exactly 2 experts)
#pragma unroll
  for (int mf = 0; mf < 2; ++mf) {
#pragma unroll
    for (int nf = 0; nf < 4; ++nf) {
      const int hcol = hbase + waveN * 64 + nf * 16 + l16;
#pragma unroll
      for (int r = 0; r < 8; ++r) {
        const int lm = waveM * 32 + mf * 16 + half * 8 + r;
        if (mbase + lm < c) {
          const int t = toksh[lm];
          const float w = wtsh[lm];
          atomicAdd(&out[(size_t)t * HDIM + hcol], w * acc[mf][nf][r]);
        }
      }
    }
  }
}

// ---------- host launcher ----------

extern "C" void kernel_launch(void* const* d_in, const int* in_sizes, int n_in,
                              void* d_out, int out_size, void* d_ws, size_t ws_size,
                              hipStream_t stream) {
  const float* x      = (const float*)d_in[0];  // [T, H]
  const float* gate_w = (const float*)d_in[1];  // [E, H]
  const float* w1     = (const float*)d_in[2];  // [E, I, H]
  const float* w2     = (const float*)d_in[3];  // [E, H, I]
  const float* w3     = (const float*)d_in[4];  // [E, I, H]
  float* out = (float*)d_out;                   // [T, H]

  char* ws = (char*)d_ws;
  int*   cnt  = (int*)ws;                                         // E ints
  int*   list = (int*)(ws + 256);                                 // E*T ints
  float* wgt  = (float*)(ws + 256 + (size_t)NEXP * T_TOK * 4);    // E*T floats
  unsigned short* act =
      (unsigned short*)(ws + 256 + 2 * (size_t)NEXP * T_TOK * 4); // T*I bf16

  moe_zero_kernel<<<(T_TOK * HDIM / 4) / 256, 256, 0, stream>>>(out, cnt);
  moe_router_kernel<<<T_TOK / 8, 256, 0, stream>>>(x, gate_w, cnt, list, wgt);
  for (int e = 0; e < NEXP; ++e) {
    moe_up_kernel<<<dim3(T_TOK / BM, IDIM / BN), 256, 0, stream>>>(
        x, w1, w3, cnt, list, act, e);
    moe_down_kernel<<<dim3(T_TOK / BM, HDIM / BN), 256, 0, stream>>>(
        w2, cnt, list, wgt, act, out, e);
  }
}
